// SWSeMaskBlock_8684423872854
// MI455X (gfx1250) — matrix operand order repaired and checked by reading, not compiled, not run on hardware
//
#include <hip/hip_runtime.h>
#include <hip/hip_bf16.h>

typedef __attribute__((ext_vector_type(16))) __bf16 v16bf;
typedef __attribute__((ext_vector_type(8)))  float  v8f;

#define C_DIM 1024
#define K_DIM 150
#define N_TOK 49
#define WSZ   7
#define NWIN  1024
#define L_DIM 3136
#define B_DIM 16

// workspace offsets (bytes)
#define XN_OFF     0ull                     // 50176*1024*2 = 102,760,448
#define STATS_OFF  102760448ull             // 50176*2*4    =     401,408
#define WQ_OFF     103161856ull             // 150*1024*2   =     307,200
#define WK_OFF     103469056ull             // 150*1024*2
#define WV_OFF     103776256ull             // 1024*1024*2  =   2,097,152
#define WR_OFF     105873408ull             // 1024*1024*2  -> 107,970,560 total

// dynamic LDS layout (phases don't overlap in time where regions alias)
#define SCORE_OFF  0                        // bf16 [64][64]   = 8192
#define SEGA_OFF   8192                     // bf16 [64][160]  = 20480
#define SEGB_OFF   28672                    // bf16 [64][160]  = 20480
#define LOGIT_OFF  49152                    // f32  [64][64]   = 16384
#define ATTN_OFF   8192                     // bf16 [64][1024] = 131072 (aliases segA/segB/logit)
#define VT_OFF     139264                   // bf16 8*[16][64] = 16384
#define SMEM_BYTES 155648

__device__ __forceinline__ int win_row(int w, int n) {
    // window w in [0,1024): b = w/64, window (wh,ww) in 8x8 grid; token n in [0,49)
    int b  = w >> 6;
    int wi = w & 63;
    int wh = wi >> 3, ww = wi & 7;
    int r  = n / WSZ, c = n - r * WSZ;
    return b * L_DIM + (wh * WSZ + r) * 56 + (ww * WSZ + c);
}

// A-fragment (16x32 bf16): lane holds row (lane&15); half=lane>>4 selects K in
// {16h..16h+15}, contiguous. rowptr = row base + k0.
__device__ __forceinline__ v16bf load_a_frag(const __bf16* rowptr, int half) {
    v16bf a;
#pragma unroll
    for (int v = 0; v < 8; ++v) {
        int koff = half * 16 + 2 * v;
        a[2 * v]     = rowptr[koff];
        a[2 * v + 1] = rowptr[koff + 1];
    }
    return a;
}

// B-fragment (32x16 bf16): lane holds column (lane&15); half selects K in
// {16h..16h+15}, contiguous. colptr = col base + k0 + half*16.
__device__ __forceinline__ v16bf load_b_frag(const __bf16* colptr) {
    v16bf b;
#pragma unroll
    for (int v = 0; v < 8; ++v) {
        b[2 * v]     = colptr[2 * v];
        b[2 * v + 1] = colptr[2 * v + 1];
    }
    return b;
}

__device__ __forceinline__ v8f wmma_bf16(v16bf a, v16bf b, v8f c) {
    return __builtin_amdgcn_wmma_f32_16x16x32_bf16(false, a, false, b, (short)0, c, false, false);
}

__device__ __forceinline__ v8f zero8() {
    v8f z = {0.f, 0.f, 0.f, 0.f, 0.f, 0.f, 0.f, 0.f};
    return z;
}

// ---------------- Kernel 1: LayerNorm -> bf16 activations + (mu, rstd) ----------------
__global__ __launch_bounds__(256) void layernorm_kernel(
    const float* __restrict__ x, const float* __restrict__ nw, const float* __restrict__ nb,
    __bf16* __restrict__ xnbf, float* __restrict__ stats) {
    __shared__ float red[256];
    __shared__ float red2[256];
    const int row = blockIdx.x;
    const int tid = threadIdx.x;
    const float* xr = x + (size_t)row * C_DIM;
    float v0 = xr[tid], v1 = xr[tid + 256], v2 = xr[tid + 512], v3 = xr[tid + 768];
    red[tid]  = v0 + v1 + v2 + v3;
    red2[tid] = v0 * v0 + v1 * v1 + v2 * v2 + v3 * v3;
    __syncthreads();
    for (int off = 128; off > 0; off >>= 1) {
        if (tid < off) { red[tid] += red[tid + off]; red2[tid] += red2[tid + off]; }
        __syncthreads();
    }
    float mu  = red[0] * (1.f / 1024.f);
    float var = red2[0] * (1.f / 1024.f) - mu * mu;
    float rs  = rsqrtf(var + 1e-5f);
    if (tid == 0) { stats[2 * row] = mu; stats[2 * row + 1] = rs; }
    __bf16* orow = xnbf + (size_t)row * C_DIM;
    orow[tid]       = (__bf16)((v0 - mu) * rs * nw[tid]       + nb[tid]);
    orow[tid + 256] = (__bf16)((v1 - mu) * rs * nw[tid + 256] + nb[tid + 256]);
    orow[tid + 512] = (__bf16)((v2 - mu) * rs * nw[tid + 512] + nb[tid + 512]);
    orow[tid + 768] = (__bf16)((v3 - mu) * rs * nw[tid + 768] + nb[tid + 768]);
}

// ---------------- Kernel 2: f32 -> bf16 weight conversion ----------------
__global__ void cvt_bf16_kernel(const float* __restrict__ src, __bf16* __restrict__ dst, int n) {
    int i = blockIdx.x * blockDim.x + threadIdx.x;
    int stride = gridDim.x * blockDim.x;
    for (; i < n; i += stride) dst[i] = (__bf16)src[i];
}

// ---------------- Kernel 3: fused per-window attention ----------------
__global__ __launch_bounds__(256) void window_attn_kernel(
    const float* __restrict__ x,
    const float* __restrict__ norm_w, const float* __restrict__ norm_b,
    const float* __restrict__ bq, const float* __restrict__ bk,
    const float* __restrict__ bv, const float* __restrict__ br,
    const float* __restrict__ gamma,
    const __bf16* __restrict__ xnbf, const float* __restrict__ stats,
    const __bf16* __restrict__ wqbf, const __bf16* __restrict__ wkbf,
    const __bf16* __restrict__ wvbf, const __bf16* __restrict__ wrbf,
    float* __restrict__ sem_out, float* __restrict__ x_out) {
    extern __shared__ char smem[];
    __bf16* score  = (__bf16*)(smem + SCORE_OFF);
    __bf16* segA   = (__bf16*)(smem + SEGA_OFF);
    __bf16* segB   = (__bf16*)(smem + SEGB_OFF);
    float*  logit  = (float*)(smem + LOGIT_OFF);
    __bf16* attn   = (__bf16*)(smem + ATTN_OFF);
    __bf16* vtiles = (__bf16*)(smem + VT_OFF);

    const int w    = blockIdx.x;
    const int tid  = threadIdx.x;
    const int wave = tid >> 5;
    const int lane = tid & 31;
    const int half = lane >> 4;
    const int lr   = lane & 15;

    // ---- Phase S: seg_map = xn@wq^T+bq, seg_ft = xn@wk^T+bk (M=64, N=160, K=1024) ----
    // wave -> (mat, mt); N-tiles processed in 2 groups of 5 sharing one A fragment/k-step.
    {
        const int mat = wave >> 2;            // 0 -> wq/seg_map, 1 -> wk/seg_ft
        const int mt  = wave & 3;
        const __bf16* wmat = mat ? wkbf : wqbf;
        const float*  bias = mat ? bk : bq;
        __bf16*       seg  = mat ? segB : segA;

        int m    = mt * 16 + lr;
        int grow = win_row(w, m < N_TOK ? m : N_TOK - 1);
        const __bf16* arow = xnbf + (size_t)grow * C_DIM;

        for (int g = 0; g < 2; ++g) {
            const __bf16* bp[5];
            v8f acc[5];
#pragma unroll
            for (int j = 0; j < 5; ++j) {
                int n  = (g * 5 + j) * 16 + lr;
                int nc = n < K_DIM ? n : K_DIM - 1;
                bp[j]  = wmat + (size_t)nc * C_DIM + half * 16;
                acc[j] = zero8();
                __builtin_prefetch(bp[j], 0, 1);
            }
#pragma unroll 2
            for (int k0 = 0; k0 < C_DIM; k0 += 32) {
                v16bf a = load_a_frag(arow + k0, half);
#pragma unroll
                for (int j = 0; j < 5; ++j)
                    acc[j] = wmma_bf16(a, load_b_frag(bp[j] + k0), acc[j]);
            }
#pragma unroll
            for (int j = 0; j < 5; ++j) {
                int cn = (g * 5 + j) * 16 + lr;
                float bval = (cn < K_DIM) ? bias[cn] : 0.f;
#pragma unroll
                for (int v = 0; v < 8; ++v) {
                    int mm = mt * 16 + v + 8 * half;
                    float val = acc[j][v] + bval;
                    seg[mm * 160 + cn] = (cn < K_DIM) ? (__bf16)val : (__bf16)0.f; // zero K-pad
                    if (mat == 0 && mm < N_TOK && cn < K_DIM)
                        sem_out[(size_t)win_row(w, mm) * K_DIM + cn] = val;
                }
            }
        }
    }
    __syncthreads();

    // ---- Phase L: logits = seg_map @ seg_ft^T (M=64, N=64, K=160) ----
    for (int i = 0; i < 2; ++i) {
        int t  = wave * 2 + i;            // 16 tiles
        int mt = t >> 2, nt = t & 3;
        const __bf16* arow = segA + (mt * 16 + lr) * 160;
        const __bf16* bcol = segB + (nt * 16 + lr) * 160 + half * 16;
        v8f acc = zero8();
#pragma unroll
        for (int k0 = 0; k0 < 160; k0 += 32)
            acc = wmma_bf16(load_a_frag(arow + k0, half), load_b_frag(bcol + k0), acc);
        int cn = nt * 16 + lr;
#pragma unroll
        for (int v = 0; v < 8; ++v) logit[(mt * 16 + v + 8 * half) * 64 + cn] = acc[v];
    }
    __syncthreads();

    // ---- Phase SM: row softmax (49 valid rows/cols), zero-padded bf16 score ----
    if (tid < 64) {
        __bf16* srow = score + tid * 64;
        if (tid < N_TOK) {
            const float* lg = logit + tid * 64;
            float mx = -3.4e38f;
            for (int j = 0; j < N_TOK; ++j) mx = fmaxf(mx, lg[j]);
            float s = 0.f;
            for (int j = 0; j < N_TOK; ++j) s += __expf(lg[j] - mx);
            float inv = 1.f / s;
            for (int j = 0; j < 64; ++j)
                srow[j] = (j < N_TOK) ? (__bf16)(__expf(lg[j] - mx) * inv) : (__bf16)0.f;
        } else {
            for (int j = 0; j < 64; ++j) srow[j] = (__bf16)0.f;
        }
    }
    __syncthreads();

    // ---- Phase AV: V = xn@wv^T+bv per 16-col tile, then attn = score@V ----
    __bf16* vt = vtiles + wave * (16 * 64);   // V tile transposed: vt[n_local][token]
    for (int iter = 0; iter < 8; ++iter) {
        int colbase = iter * 128 + wave * 16;
        const __bf16* bcol = wvbf + (size_t)(colbase + lr) * C_DIM + half * 16;
        __builtin_prefetch(bcol, 0, 1);
        // V block: share the B (wv) fragment across the 4 M-tiles each k-step.
        {
            const __bf16* ap[4];
            v8f acc[4];
#pragma unroll
            for (int mt = 0; mt < 4; ++mt) {
                int m    = mt * 16 + lr;
                int grow = win_row(w, m < N_TOK ? m : N_TOK - 1);
                ap[mt]   = xnbf + (size_t)grow * C_DIM;
                acc[mt]  = zero8();
            }
#pragma unroll 2
            for (int k0 = 0; k0 < C_DIM; k0 += 32) {
                v16bf b = load_b_frag(bcol + k0);
#pragma unroll
                for (int mt = 0; mt < 4; ++mt)
                    acc[mt] = wmma_bf16(load_a_frag(ap[mt] + k0, half), b, acc[mt]);
            }
            float bval = bv[colbase + lr];
#pragma unroll
            for (int mt = 0; mt < 4; ++mt) {
#pragma unroll
                for (int v = 0; v < 8; ++v) {
                    int mm = mt * 16 + v + 8 * half;
                    vt[lr * 64 + mm] = (mm < N_TOK) ? (__bf16)(acc[mt][v] + bval) : (__bf16)0.f;
                }
            }
        }
        __syncthreads();
#pragma unroll
        for (int mt = 0; mt < 4; ++mt) {
            const __bf16* arow  = score + (mt * 16 + lr) * 64;
            const __bf16* bcol2 = vt + lr * 64 + half * 16;
            v8f acc = zero8();
#pragma unroll
            for (int k0 = 0; k0 < 64; k0 += 32)
                acc = wmma_bf16(load_a_frag(arow + k0, half), load_b_frag(bcol2 + k0), acc);
#pragma unroll
            for (int v = 0; v < 8; ++v) {
                int mm = mt * 16 + v + 8 * half;
                attn[mm * 1024 + colbase + lr] = (__bf16)acc[v];
            }
        }
        __syncthreads();
    }

    // ---- Phase R: out = gamma*(attn@wr^T + br) + win (win recomputed in f32) ----
    // wave -> (mt, ntg); N-tiles in 8 groups of 4 sharing one LDS A fragment/k-step.
    float gma = gamma[0];
    {
        const int mt  = wave & 3;
        const int ntg = wave >> 2;            // 0..1, each covers 32 N-tiles
        const __bf16* arow = attn + (mt * 16 + lr) * 1024;
        for (int g = 0; g < 8; ++g) {
            int ntbase = ntg * 32 + g * 4;
            const __bf16* bp[4];
            v8f acc[4];
#pragma unroll
            for (int j = 0; j < 4; ++j) {
                bp[j]  = wrbf + (size_t)((ntbase + j) * 16 + lr) * C_DIM + half * 16;
                acc[j] = zero8();
                __builtin_prefetch(bp[j], 0, 1);
            }
#pragma unroll 2
            for (int k0 = 0; k0 < C_DIM; k0 += 32) {
                v16bf a = load_a_frag(arow + k0, half);
#pragma unroll
                for (int j = 0; j < 4; ++j)
                    acc[j] = wmma_bf16(a, load_b_frag(bp[j] + k0), acc[j]);
            }
#pragma unroll
            for (int j = 0; j < 4; ++j) {
                int cn = (ntbase + j) * 16 + lr;
                float brv = br[cn], nwv = norm_w[cn], nbv = norm_b[cn];
#pragma unroll
                for (int v = 0; v < 8; ++v) {
                    int mm = mt * 16 + v + 8 * half;
                    if (mm < N_TOK) {
                        int g2 = win_row(w, mm);
                        float mu = stats[2 * g2];
                        float rs = stats[2 * g2 + 1];
                        float winv = (x[(size_t)g2 * C_DIM + cn] - mu) * rs * nwv + nbv;
                        x_out[(size_t)g2 * C_DIM + cn] = gma * (acc[j][v] + brv) + winv;
                    }
                }
            }
        }
    }
}

extern "C" void kernel_launch(void* const* d_in, const int* in_sizes, int n_in,
                              void* d_out, int out_size, void* d_ws, size_t ws_size,
                              hipStream_t stream) {
    (void)in_sizes; (void)n_in; (void)out_size; (void)ws_size;
    const float* x      = (const float*)d_in[0];
    const float* norm_w = (const float*)d_in[1];
    const float* norm_b = (const float*)d_in[2];
    const float* wq     = (const float*)d_in[3];
    const float* bq     = (const float*)d_in[4];
    const float* wk     = (const float*)d_in[5];
    const float* bk     = (const float*)d_in[6];
    const float* wv     = (const float*)d_in[7];
    const float* bv     = (const float*)d_in[8];
    const float* wr     = (const float*)d_in[9];
    const float* br     = (const float*)d_in[10];
    const float* gamma  = (const float*)d_in[11];

    char* ws = (char*)d_ws;
    __bf16* xnbf  = (__bf16*)(ws + XN_OFF);
    float*  stats = (float*)(ws + STATS_OFF);
    __bf16* wqbf  = (__bf16*)(ws + WQ_OFF);
    __bf16* wkbf  = (__bf16*)(ws + WK_OFF);
    __bf16* wvbf  = (__bf16*)(ws + WV_OFF);
    __bf16* wrbf  = (__bf16*)(ws + WR_OFF);

    float* sem_out = (float*)d_out;
    float* x_out   = sem_out + (size_t)B_DIM * L_DIM * K_DIM;

    layernorm_kernel<<<B_DIM * L_DIM, 256, 0, stream>>>(x, norm_w, norm_b, xnbf, stats);
    cvt_bf16_kernel<<<512, 256, 0, stream>>>(wq, wqbf, K_DIM * C_DIM);
    cvt_bf16_kernel<<<512, 256, 0, stream>>>(wk, wkbf, K_DIM * C_DIM);
    cvt_bf16_kernel<<<2048, 256, 0, stream>>>(wv, wvbf, C_DIM * C_DIM);
    cvt_bf16_kernel<<<2048, 256, 0, stream>>>(wr, wrbf, C_DIM * C_DIM);

    (void)hipFuncSetAttribute(reinterpret_cast<const void*>(window_attn_kernel),
                              hipFuncAttributeMaxDynamicSharedMemorySize, SMEM_BYTES);
    window_attn_kernel<<<NWIN, 256, SMEM_BYTES, stream>>>(
        x, norm_w, norm_b, bq, bk, bv, br, gamma,
        xnbf, stats, wqbf, wkbf, wvbf, wrbf, sem_out, x_out);
}
